// SimpleCrossAttention_54692113547437
// MI455X (gfx1250) — matrix operand order repaired and checked by reading, not compiled, not run on hardware
//
#include <hip/hip_runtime.h>
#include <hip/hip_bf16.h>

typedef __attribute__((ext_vector_type(16))) __bf16 v16bf;
typedef __attribute__((ext_vector_type(8)))  __bf16 v8bf;
typedef __attribute__((ext_vector_type(8)))  float  v8f;
typedef __attribute__((ext_vector_type(4)))  float  v4f;

#define N_HEADS 8
#define HD      64
#define CDIM    512
#define NLTOT   2048
#define ROWS    4096   // B * NL

// ---------------------------------------------------------------------------
// Kernel 1: projections. Y[p] = X @ W[p]^T  (4096x512 @ 512x512), bf16 output.
// alpha = (1/sqrt(C))/tau folded into the Q projections (p = 0, 2).
// ws layout: [Q1 | K1 | Q2 | K2], each 4096*512 bf16.
// Block = 128 threads (4 waves); wave computes 16 rows x 64 cols.
// ---------------------------------------------------------------------------
__global__ __launch_bounds__(128)
void proj_kernel(const float* __restrict__ f1, const float* __restrict__ f2,
                 const float* __restrict__ Wq1, const float* __restrict__ Wk1,
                 const float* __restrict__ Wq2, const float* __restrict__ Wk2,
                 const float* __restrict__ tauPtr, __bf16* __restrict__ ws)
{
    int gid = blockIdx.x;
    int ct  = gid & 7;          // col tile of 64
    int mt  = (gid >> 3) & 63;  // row tile of 64
    int p   = gid >> 9;         // projection 0..3

    const float* X = (p < 2) ? f1 : f2;
    const float* W = (p == 0) ? Wq1 : (p == 1) ? Wk1 : (p == 2) ? Wq2 : Wk2;
    __bf16* Y = ws + (size_t)p * ROWS * CDIM;
    float alpha = ((p & 1) == 0) ? (0.04419417382415922f / tauPtr[0]) : 1.0f;

    int lane = threadIdx.x & 31;
    int wv   = threadIdx.x >> 5;
    int half = lane >> 4;
    int l15  = lane & 15;

    int rowA = mt * 64 + wv * 16 + l15;     // A-matrix: M = lane%16
    v8f acc[4] = {};

    for (int kb = 0; kb < CDIM; kb += 32) {
        // ---- A fragment (16x32 bf16): contiguous K order, slot s holds K = kb+16h+s
        union { v16bf v; v8bf h[2]; } a;
        {
            const float* xr = X + (size_t)rowA * CDIM + kb;
            const v4f* x0 = (const v4f*)(xr + 16 * half);       // K = kb+16h+0..7
            const v4f* x1 = (const v4f*)(xr + 8 + 16 * half);   // K = kb+16h+8..15
            v4f c0 = x0[0], c1 = x0[1], c2 = x1[0], c3 = x1[1];
            #pragma unroll
            for (int s = 0; s < 4; ++s) {
                a.h[0][s]     = (__bf16)c0[s];
                a.h[0][s + 4] = (__bf16)c1[s];
                a.h[1][s]     = (__bf16)c2[s];
                a.h[1][s + 4] = (__bf16)c3[s];
            }
        }
        // ---- B fragments: B[k=c][n=dout] = W[dout][c]; lane%16=N, slots=K+16h
        #pragma unroll
        for (int j = 0; j < 4; ++j) {
            int dout = ct * 64 + j * 16 + l15;
            const v4f* wp = (const v4f*)(W + (size_t)dout * CDIM + kb + 16 * half);
            v4f w0 = wp[0], w1 = wp[1], w2 = wp[2], w3 = wp[3];
            v16bf bm;
            #pragma unroll
            for (int s = 0; s < 4; ++s) {
                bm[s]      = (__bf16)w0[s];
                bm[s + 4]  = (__bf16)w1[s];
                bm[s + 8]  = (__bf16)w2[s];
                bm[s + 12] = (__bf16)w3[s];
            }
            acc[j] = __builtin_amdgcn_wmma_f32_16x16x32_bf16(
                false, a.v, false, bm, (short)0, acc[j], false, false);
        }
    }

    // ---- epilogue: D layout M = r + 8*(lane/16), N = lane%16
    int rowD = mt * 64 + wv * 16 + 8 * half;
    #pragma unroll
    for (int j = 0; j < 4; ++j) {
        int col = ct * 64 + j * 16 + l15;
        #pragma unroll
        for (int r = 0; r < 8; ++r)
            Y[(size_t)(rowD + r) * CDIM + col] = (__bf16)(acc[j][r] * alpha);
    }
}

// ---------------------------------------------------------------------------
// Kernel 2: fused scores + softmax + block-max pooling.
// One block = one (dir, b, h, query-tile-of-64). 4 waves; wave owns 16 queries.
// K tiles (32 keys x 64 hd bf16 = 4 KB) are staged global->LDS ONCE per block
// with async-to-LDS copies (ASYNCcnt), double-buffered so the copy of tile n+1
// overlaps the WMMAs + online softmax of tile n. B-fragments come from LDS.
// ---------------------------------------------------------------------------
__global__ __launch_bounds__(128)
void attn_pool_kernel(const __bf16* __restrict__ ws, float* __restrict__ out)
{
    __shared__ float  ldsM[64][65];        // M_qn per (query-in-tile, key-block)
    __shared__ float  ldsm[64];            // global row max
    __shared__ float  ldsrz[64];           // 1/Z
    __shared__ __bf16 ktile[2][32][64];    // double-buffered K tile (2 x 4 KB)

    int gid = blockIdx.x;
    int qt  = gid & 31;
    int h   = (gid >> 5) & 7;
    int b   = (gid >> 8) & 1;
    int dir = gid >> 9;

    const __bf16* Qb = ws + (size_t)(dir == 0 ? 0 : 2) * ROWS * CDIM; // Q1 / Q2
    const __bf16* Kb = ws + (size_t)(dir == 0 ? 3 : 1) * ROWS * CDIM; // K2 / K1

    int tid  = threadIdx.x;
    int lane = tid & 31;
    int wv   = tid >> 5;
    int half = lane >> 4;
    int l15  = lane & 15;

    // ---- load the wave's A fragments (16 queries x 64 hd), d-chunks {0,32}
    union { v16bf v; v8bf hh[2]; } a0, a1;
    {
        const __bf16* qr = Qb + ((size_t)(b * NLTOT + qt * 64 + wv * 16 + l15)) * CDIM + h * HD;
        a0.hh[0] = *(const v8bf*)(qr + 0  + 16 * half);
        a0.hh[1] = *(const v8bf*)(qr + 8  + 16 * half);
        a1.hh[0] = *(const v8bf*)(qr + 32 + 16 * half);
        a1.hh[1] = *(const v8bf*)(qr + 40 + 16 * half);
    }

    // async copy of one K tile: 32 rows x 128 B = 256 x 16 B; 128 thr x 2 each
    unsigned ldsK[2] = { (unsigned)(size_t)&ktile[0][0][0],
                         (unsigned)(size_t)&ktile[1][0][0] };
    size_t headOff = (size_t)(b * NLTOT) * CDIM + h * HD;
    auto issue_copy = [&](int n, int buf) {
        #pragma unroll
        for (int j = 0; j < 2; ++j) {
            int i     = tid + 128 * j;
            int row   = i >> 3;            // key within tile
            int chunk = i & 7;             // 16-byte chunk within 128-B row
            const __bf16* g = Kb + headOff + (size_t)(n * 32 + row) * CDIM + chunk * 8;
            unsigned l = ldsK[buf] + (unsigned)(row * 128 + chunk * 16);
            asm volatile("global_load_async_to_lds_b128 %0, %1, off"
                         :: "v"(l), "v"(g) : "memory");
        }
    };

    float mrun[8], zrun[8];
    #pragma unroll
    for (int r = 0; r < 8; ++r) { mrun[r] = -__builtin_inff(); zrun[r] = 0.0f; }

    int qloc0 = wv * 16 + 8 * half;   // D-row base for this lane

    // prologue: stage tile 0
    issue_copy(0, 0);
    asm volatile("s_wait_asynccnt 0x0" ::: "memory");
    __syncthreads();

    for (int n = 0; n < 64; ++n) {
        if (n + 1 < 64) issue_copy(n + 1, (n + 1) & 1);

        const __bf16* kt = &ktile[n & 1][0][0];
        v8f acc0 = {}, acc1 = {};
        {
            // keys 0..15 of block (N = l15), B slots: d = dbase + 16*half + s
            v16bf b0 = *(const v16bf*)(kt + l15 * 64 + 16 * half);          // d 0..31
            acc0 = __builtin_amdgcn_wmma_f32_16x16x32_bf16(
                false, a0.v, false, b0, (short)0, acc0, false, false);
            v16bf b1 = *(const v16bf*)(kt + l15 * 64 + 32 + 16 * half);     // d 32..63
            acc0 = __builtin_amdgcn_wmma_f32_16x16x32_bf16(
                false, a1.v, false, b1, (short)0, acc0, false, false);
            // keys 16..31 of block
            v16bf b2 = *(const v16bf*)(kt + (16 + l15) * 64 + 16 * half);
            acc1 = __builtin_amdgcn_wmma_f32_16x16x32_bf16(
                false, a0.v, false, b2, (short)0, acc1, false, false);
            v16bf b3 = *(const v16bf*)(kt + (16 + l15) * 64 + 32 + 16 * half);
            acc1 = __builtin_amdgcn_wmma_f32_16x16x32_bf16(
                false, a1.v, false, b3, (short)0, acc1, false, false);
        }

        float t[8], e[8];
        #pragma unroll
        for (int r = 0; r < 8; ++r) t[r] = fmaxf(acc0[r], acc1[r]);
        #pragma unroll
        for (int r = 0; r < 8; ++r) {
            #pragma unroll
            for (int m = 1; m <= 8; m <<= 1)
                t[r] = fmaxf(t[r], __shfl_xor(t[r], m, 32));   // 16-lane butterfly
        }
        #pragma unroll
        for (int r = 0; r < 8; ++r)
            e[r] = __expf(acc0[r] - t[r]) + __expf(acc1[r] - t[r]);
        #pragma unroll
        for (int r = 0; r < 8; ++r) {
            #pragma unroll
            for (int m = 1; m <= 8; m <<= 1)
                e[r] += __shfl_xor(e[r], m, 32);
        }
        #pragma unroll
        for (int r = 0; r < 8; ++r) {
            float nm = fmaxf(mrun[r], t[r]);
            zrun[r] = zrun[r] * __expf(mrun[r] - nm) + e[r] * __expf(t[r] - nm);
            mrun[r] = nm;
        }
        if (l15 == 0) {
            #pragma unroll
            for (int r = 0; r < 8; ++r) ldsM[qloc0 + r][n] = t[r];
        }

        // tile n+1 resident + all waves done with buffer n&1 before reuse
        asm volatile("s_wait_asynccnt 0x0" ::: "memory");
        __syncthreads();
    }

    if (l15 == 0) {
        #pragma unroll
        for (int r = 0; r < 8; ++r) {
            ldsm[qloc0 + r]  = mrun[r];
            ldsrz[qloc0 + r] = 1.0f / zrun[r];
        }
    }
    __syncthreads();

    // ---- finalize: pooled(q,n) = exp(M - m)/Z; mean over l_q, heads, dirs
    int nk = tid & 63;
    int qg = tid >> 6;   // which group of 32 queries (output row)
    float s = 0.0f;
    #pragma unroll 4
    for (int lq = 0; lq < 32; ++lq) {
        int ql = qg * 32 + lq;
        s += __expf(ldsM[ql][nk] - ldsm[ql]) * ldsrz[ql];
    }
    // weight = 1 / (L * H * 2 dirs) = 1/512
    atomicAdd(&out[(size_t)b * 4096 + (qt * 2 + qg) * 64 + nk], s * (1.0f / 512.0f));
}

__global__ void zero_kernel(float* __restrict__ out, int n)
{
    int i = blockIdx.x * blockDim.x + threadIdx.x;
    if (i < n) out[i] = 0.0f;
}

extern "C" void kernel_launch(void* const* d_in, const int* in_sizes, int n_in,
                              void* d_out, int out_size, void* d_ws, size_t ws_size,
                              hipStream_t stream)
{
    const float* f1  = (const float*)d_in[0];
    const float* f2  = (const float*)d_in[1];
    const float* Wq1 = (const float*)d_in[2];
    const float* Wk1 = (const float*)d_in[3];
    const float* Wq2 = (const float*)d_in[4];
    const float* Wk2 = (const float*)d_in[5];
    const float* tau = (const float*)d_in[6];
    __bf16* ws  = (__bf16*)d_ws;   // needs 4 * 4096 * 512 * 2B = 16 MB
    float*  out = (float*)d_out;

    // 4 projections x 64 row-tiles x 8 col-tiles
    proj_kernel<<<2048, 128, 0, stream>>>(f1, f2, Wq1, Wk1, Wq2, Wk2, tau, ws);
    zero_kernel<<<(out_size + 255) / 256, 256, 0, stream>>>(out, out_size);
    // 2 dirs x 2 batch x 8 heads x 32 query-tiles
    attn_pool_kernel<<<1024, 128, 0, stream>>>(ws, out);
}
